// MultiHeadAttention_37512244363833
// MI455X (gfx1250) — matrix operand order repaired and checked by reading, not compiled, not run on hardware
//
#include <hip/hip_runtime.h>
#include <hip/hip_bf16.h>
#include <math.h>

#define D_MODEL 1024
#define N_HEADS 16
#define DK      64
#define BATCH   2
#define SEQ     2048

typedef __attribute__((ext_vector_type(16))) __bf16 bf16x16;
typedef __attribute__((ext_vector_type(8)))  __bf16 bf16x8;
typedef __attribute__((ext_vector_type(4)))  __bf16 bf16x4;
typedef __attribute__((ext_vector_type(4)))  float  f32x4;
typedef __attribute__((ext_vector_type(8)))  float  f32x8;

// ---------------------------------------------------------------- helpers

static __device__ __forceinline__ __bf16 to_bf16(float f) {
    union { float f; unsigned u; } v; v.f = f;
    unsigned r = v.u + 0x7FFFu + ((v.u >> 16) & 1u);   // round-to-nearest-even
    unsigned short h = (unsigned short)(r >> 16);
    return __builtin_bit_cast(__bf16, h);
}

// ---- 16-lane row reductions via DPP16 (VALU, co-executes with WMMA).
// Order matters: after xor1/xor2 each quad is uniform, so row_half_mirror
// (lane ^ 7) acts as xor4 and row_mirror (lane ^ 15) acts as xor8.
#if __has_builtin(__builtin_amdgcn_update_dpp)
template <int CTRL>
static __device__ __forceinline__ float dpp_mov(float x) {
    return __builtin_bit_cast(float,
        __builtin_amdgcn_update_dpp(0, __builtin_bit_cast(int, x), CTRL, 0xF, 0xF, true));
}
static __device__ __forceinline__ float row16_max(float x) {
    x = fmaxf(x, dpp_mov<0xB1>(x));     // quad_perm [1,0,3,2]
    x = fmaxf(x, dpp_mov<0x4E>(x));     // quad_perm [2,3,0,1]
    x = fmaxf(x, dpp_mov<0x141>(x));    // row_half_mirror
    x = fmaxf(x, dpp_mov<0x140>(x));    // row_mirror
    return x;
}
static __device__ __forceinline__ float row16_sum(float x) {
    x += dpp_mov<0xB1>(x);
    x += dpp_mov<0x4E>(x);
    x += dpp_mov<0x141>(x);
    x += dpp_mov<0x140>(x);
    return x;
}
#else
static __device__ __forceinline__ float row16_max(float x) {
    x = fmaxf(x, __shfl_xor(x, 1, 32));
    x = fmaxf(x, __shfl_xor(x, 2, 32));
    x = fmaxf(x, __shfl_xor(x, 4, 32));
    x = fmaxf(x, __shfl_xor(x, 8, 32));
    return x;
}
static __device__ __forceinline__ float row16_sum(float x) {
    x += __shfl_xor(x, 1, 32);
    x += __shfl_xor(x, 2, 32);
    x += __shfl_xor(x, 4, 32);
    x += __shfl_xor(x, 8, 32);
    return x;
}
#endif

// A-fragment (16x32 bf16, ISA 7.12.2 16-bit A layout).
// base points at element (row0, k0) of a row-major bf16 matrix, ld = row stride.
// lanes 0-15: row = lane,    K = {k0+0..7,  k0+16..23}
// lanes16-31: row = lane-16, K = {k0+8..15, k0+24..31}
static __device__ __forceinline__ bf16x16 load_a_frag(const __bf16* base, int ld, int lane) {
    int half = lane >> 4;
    int m    = lane & 15;
    int kb   = half << 3;                       // 0 or 8
    const __bf16* p = base + (size_t)m * ld + kb;
    bf16x8 a0 = *(const bf16x8*)(p);
    bf16x8 a1 = *(const bf16x8*)(p + 16);
    bf16x16 r;
#pragma unroll
    for (int i = 0; i < 8; ++i) { r[i] = a0[i]; r[i + 8] = a1[i]; }
    return r;
}

// B-fragment (32x16 bf16). B[k][n] read from a source where each column n is
// contiguous in k (source row n holds k contiguously), ld = that stride.
// lanes 0-15: col = lane,    K = {k0+0..7,  k0+16..23}
// lanes16-31: col = lane-16, K = {k0+8..15, k0+24..31}
static __device__ __forceinline__ bf16x16 load_b_frag(const __bf16* base, int ld, int lane) {
    int half = lane >> 4;
    int n    = lane & 15;
    const __bf16* p = base + (size_t)n * ld + (half << 3);
    bf16x8 b0 = *(const bf16x8*)(p);
    bf16x8 b1 = *(const bf16x8*)(p + 16);
    bf16x16 r;
#pragma unroll
    for (int i = 0; i < 8; ++i) { r[i] = b0[i]; r[i + 8] = b1[i]; }
    return r;
}

static __device__ __forceinline__ f32x8 wmma_bf16(bf16x16 a, bf16x16 b, f32x8 c) {
    return __builtin_amdgcn_wmma_f32_16x16x32_bf16(false, a, false, b, (short)0, c,
                                                   false, false);
}

// ---------------------------------------------------------------- convert (4-wide)

__global__ void cvt_f32_bf16_v4(const float* __restrict__ in, __bf16* __restrict__ out,
                                int n4) {
    int i = blockIdx.x * blockDim.x + threadIdx.x;
    if (i < n4) {
        f32x4 v = *(const f32x4*)(in + 4 * (size_t)i);
        bf16x4 o;
#pragma unroll
        for (int j = 0; j < 4; ++j) o[j] = to_bf16(v[j]);
        *(bf16x4*)(out + 4 * (size_t)i) = o;
    }
}

// ---------------------------------------------------------------- GEMM  Y = (X*W^T + b)*s
// X: [M,K] bf16 row-major, W: [N,K] bf16 row-major (torch Linear weight), bias f32[N].
// 128 threads = 4 waves; block tile 64x128; wave tile 32x64 (2x4 WMMA accumulators).
template <bool OUT_BF16>
__global__ __launch_bounds__(128, 1)
void gemm_xwt_wmma(const __bf16* __restrict__ X, const __bf16* __restrict__ W,
                   const float* __restrict__ bias, void* __restrict__ out,
                   int M, int N, int K, float oscale) {
    const int lane = threadIdx.x & 31;
    const int wave = threadIdx.x >> 5;
    const int m0 = blockIdx.x * 64 + (wave >> 1) * 32;
    const int n0 = blockIdx.y * 128 + (wave & 1) * 64;

    f32x8 acc[2][4];
#pragma unroll
    for (int i = 0; i < 2; ++i)
#pragma unroll
        for (int j = 0; j < 4; ++j) acc[i][j] = f32x8{};

    for (int k = 0; k < K; k += 32) {
        bf16x16 a0 = load_a_frag(X + (size_t)m0 * K + k, K, lane);
        bf16x16 a1 = load_a_frag(X + (size_t)(m0 + 16) * K + k, K, lane);
        bf16x16 b[4];
#pragma unroll
        for (int j = 0; j < 4; ++j)
            b[j] = load_b_frag(W + (size_t)(n0 + 16 * j) * K + k, K, lane);
#pragma unroll
        for (int j = 0; j < 4; ++j) {
            acc[0][j] = wmma_bf16(a0, b[j], acc[0][j]);
            acc[1][j] = wmma_bf16(a1, b[j], acc[1][j]);
        }
    }

    // C-fragment layout: VGPR r -> row (r | r+8), lanes 0-15 / 16-31 -> cols 0-15.
    const int half = lane >> 4;
    const int nl   = lane & 15;
#pragma unroll
    for (int j = 0; j < 4; ++j) {
        int n = n0 + 16 * j + nl;
        float bj = bias[n];
#pragma unroll
        for (int mi = 0; mi < 2; ++mi) {
#pragma unroll
            for (int r = 0; r < 8; ++r) {
                int m = m0 + 16 * mi + r + (half << 3);
                float v = (acc[mi][j][r] + bj) * oscale;
                if (OUT_BF16) ((__bf16*)out)[(size_t)m * N + n] = to_bf16(v);
                else          ((float*)out)[(size_t)m * N + n] = v;
            }
        }
    }
}

// ---------------------------------------------------------------- flash attention
// grid: (S/64, H, B); block: 128 threads (4 waves). Wave w owns query rows
// [q0 + 16w, q0 + 16w + 15]. Key loop in 64-key tiles (2 x K=32 WMMA chunks),
// causal, one online-softmax round per 64 keys.
// Q arrives pre-scaled by 1/sqrt(dk) (folded into the Q projection).
__global__ __launch_bounds__(128, 1)
void flash_attn_wmma(const __bf16* __restrict__ Qp, const __bf16* __restrict__ Kp,
                     const __bf16* __restrict__ Vp, __bf16* __restrict__ Hcat) {
    const int S = SEQ, D = D_MODEL;
    const int lane = threadIdx.x & 31;
    const int wave = threadIdx.x >> 5;
    const int half = lane >> 4;
    const int nl   = lane & 15;

    const int q0 = blockIdx.x * 64;
    const int h  = blockIdx.y;
    const int b  = blockIdx.z;
    const int mrow0 = q0 + wave * 16;

    const __bf16* Qb = Qp + (size_t)b * S * D + h * DK;
    const __bf16* Kb = Kp + (size_t)b * S * D + h * DK;
    const __bf16* Vb = Vp + (size_t)b * S * D + h * DK;

    __shared__ __bf16 Vt[DK * 64];        // transposed V tile: [dk][key], 8 KB
    __shared__ __bf16 Pb[4][16 * 64];     // per-wave P staging, 2 KB each

    // Q fragments for this wave's 16 rows (dk = 64 -> two 16x32 A-fragments), kept live.
    bf16x16 qa0 = load_a_frag(Qb + (size_t)mrow0 * D + 0,  D, lane);
    bf16x16 qa1 = load_a_frag(Qb + (size_t)mrow0 * D + 32, D, lane);

    f32x8 o[4];                            // O accumulators: 16 rows x 64 dk
#pragma unroll
    for (int j = 0; j < 4; ++j) o[j] = f32x8{};
    float mrun[8], lrun[8];
#pragma unroll
    for (int r = 0; r < 8; ++r) { mrun[r] = -1e30f; lrun[r] = 0.f; }

    const int nkt = q0 / 64 + 1;          // uniform across block (causal bound)
    for (int t = 0; t < nkt; ++t) {
        const int k0 = t * 64;

        // stage V tile transposed into LDS: thread -> (key, 32-dk strip)
        {
            int key = threadIdx.x & 63;
            int dkb = (threadIdx.x >> 6) * 32;
            const __bf16* vp = Vb + (size_t)(k0 + key) * D + dkb;
            bf16x16 v0 = *(const bf16x16*)vp;
            bf16x16 v1 = *(const bf16x16*)(vp + 16);
#pragma unroll
            for (int i = 0; i < 16; ++i) {
                Vt[(dkb + i) * 64 + key]      = v0[i];
                Vt[(dkb + 16 + i) * 64 + key] = v1[i];
            }
            // prefetch next tile's K/V rows (global_prefetch_b8)
            if (t + 1 < nkt) {
                __builtin_prefetch(Vb + (size_t)(k0 + 64 + key) * D + dkb, 0, 1);
                __builtin_prefetch(Kb + (size_t)(k0 + 64 + key) * D + dkb, 0, 1);
            }
        }
        __syncthreads();

        // S = Q * K^T : 4 col-tiles of 16 keys, dk split in two K=32 chunks
        f32x8 s[4];
#pragma unroll
        for (int j = 0; j < 4; ++j) {
            const __bf16* kt = Kb + (size_t)(k0 + 16 * j) * D;
            bf16x16 kb0 = load_b_frag(kt + 0,  D, lane);
            bf16x16 kb1 = load_b_frag(kt + 32, D, lane);
            f32x8 sj{};
            sj = wmma_bf16(qa0, kb0, sj);
            sj = wmma_bf16(qa1, kb1, sj);
            s[j] = sj;
        }

        // causal mask only when this key tile can cross the wave's diagonal
        if (k0 + 63 > mrow0) {
#pragma unroll
            for (int r = 0; r < 8; ++r) {
                int row = mrow0 + r + (half << 3);
#pragma unroll
                for (int j = 0; j < 4; ++j)
                    if (k0 + 16 * j + nl > row) s[j][r] = -1e30f;
            }
        }

        // online softmax: row stats via DPP16 butterflies (VALU only)
#pragma unroll
        for (int r = 0; r < 8; ++r) {
            float mx = fmaxf(fmaxf(s[0][r], s[1][r]), fmaxf(s[2][r], s[3][r]));
            mx = row16_max(mx);
            float mnew  = fmaxf(mrun[r], mx);
            float alpha = __expf(mrun[r] - mnew);
            float e0 = __expf(s[0][r] - mnew);
            float e1 = __expf(s[1][r] - mnew);
            float e2 = __expf(s[2][r] - mnew);
            float e3 = __expf(s[3][r] - mnew);
            float rs = row16_sum((e0 + e1) + (e2 + e3));
            lrun[r] = lrun[r] * alpha + rs;
            mrun[r] = mnew;
#pragma unroll
            for (int j = 0; j < 4; ++j) o[j][r] *= alpha;
            s[0][r] = e0; s[1][r] = e1; s[2][r] = e2; s[3][r] = e3;
        }

        // re-layout P (C-frag -> A-frag) through per-wave LDS: P is 16x64
        __bf16* Pw = Pb[wave];
#pragma unroll
        for (int r = 0; r < 8; ++r) {
            int m = r + (half << 3);
#pragma unroll
            for (int j = 0; j < 4; ++j)
                Pw[m * 64 + 16 * j + nl] = to_bf16(s[j][r]);
        }
        bf16x16 pa0 = load_a_frag(Pw + 0,  64, lane);   // keys k0+0..31
        bf16x16 pa1 = load_a_frag(Pw + 32, 64, lane);   // keys k0+32..63

        // O += P * V : 4 dk col-tiles, key dim split in two K=32 chunks
#pragma unroll
        for (int j = 0; j < 4; ++j) {
            bf16x16 vb0 = load_b_frag(&Vt[(16 * j) * 64] + 0,  64, lane);
            bf16x16 vb1 = load_b_frag(&Vt[(16 * j) * 64] + 32, 64, lane);
            o[j] = wmma_bf16(pa0, vb0, o[j]);
            o[j] = wmma_bf16(pa1, vb1, o[j]);
        }
        __syncthreads();
    }

    // normalize and write Hcat[b, q, h*64 + dk] as bf16
    __bf16* Hb = Hcat + (size_t)b * S * D + h * DK;
#pragma unroll
    for (int r = 0; r < 8; ++r) {
        int m = mrow0 + r + (half << 3);
        float inv = 1.0f / lrun[r];
#pragma unroll
        for (int j = 0; j < 4; ++j)
            Hb[(size_t)m * D + 16 * j + nl] = to_bf16(o[j][r] * inv);
    }
}

// ---------------------------------------------------------------- launch

extern "C" void kernel_launch(void* const* d_in, const int* in_sizes, int n_in,
                              void* d_out, int out_size, void* d_ws, size_t ws_size,
                              hipStream_t stream) {
    (void)in_sizes; (void)n_in; (void)out_size; (void)ws_size;

    const float* in_Q = (const float*)d_in[0];
    const float* in_K = (const float*)d_in[1];
    const float* in_V = (const float*)d_in[2];
    const float* Wq   = (const float*)d_in[3];
    const float* bq   = (const float*)d_in[4];
    const float* Wk   = (const float*)d_in[5];
    const float* bk   = (const float*)d_in[6];
    const float* Wv   = (const float*)d_in[7];
    const float* bv   = (const float*)d_in[8];
    const float* Wo   = (const float*)d_in[9];
    const float* bo   = (const float*)d_in[10];

    const int M = BATCH * SEQ;          // 4096
    const int D = D_MODEL;              // 1024
    const int nAct = M * D;             // 4,194,304
    const int nW   = D * D;             // 1,048,576

    size_t off = 0;
    auto carve = [&](size_t bytes) {
        void* p = (char*)d_ws + off;
        off += (bytes + 255) & ~(size_t)255;
        return p;
    };
    __bf16* Qin = (__bf16*)carve((size_t)nAct * 2);
    __bf16* Kin = (__bf16*)carve((size_t)nAct * 2);
    __bf16* Vin = (__bf16*)carve((size_t)nAct * 2);
    __bf16* Wqb = (__bf16*)carve((size_t)nW * 2);
    __bf16* Wkb = (__bf16*)carve((size_t)nW * 2);
    __bf16* Wvb = (__bf16*)carve((size_t)nW * 2);
    __bf16* Wob = (__bf16*)carve((size_t)nW * 2);
    __bf16* Qp  = (__bf16*)carve((size_t)nAct * 2);
    __bf16* Kp  = (__bf16*)carve((size_t)nAct * 2);
    __bf16* Vp  = (__bf16*)carve((size_t)nAct * 2);
    __bf16* Hc  = (__bf16*)carve((size_t)nAct * 2);

    // 1) f32 -> bf16 conversions (vectorized x4)
    cvt_f32_bf16_v4<<<(nAct / 4 + 255) / 256, 256, 0, stream>>>(in_Q, Qin, nAct / 4);
    cvt_f32_bf16_v4<<<(nAct / 4 + 255) / 256, 256, 0, stream>>>(in_K, Kin, nAct / 4);
    cvt_f32_bf16_v4<<<(nAct / 4 + 255) / 256, 256, 0, stream>>>(in_V, Vin, nAct / 4);
    cvt_f32_bf16_v4<<<(nW / 4 + 255) / 256, 256, 0, stream>>>(Wq, Wqb, nW / 4);
    cvt_f32_bf16_v4<<<(nW / 4 + 255) / 256, 256, 0, stream>>>(Wk, Wkb, nW / 4);
    cvt_f32_bf16_v4<<<(nW / 4 + 255) / 256, 256, 0, stream>>>(Wv, Wvb, nW / 4);
    cvt_f32_bf16_v4<<<(nW / 4 + 255) / 256, 256, 0, stream>>>(Wo, Wob, nW / 4);

    // 2) projections (bf16 out, f32 accumulate + bias).
    //    Q projection folds the 1/sqrt(dk) = 0.125 attention scale.
    dim3 gemmGrid(M / 64, D / 128);     // (64, 8)
    gemm_xwt_wmma<true><<<gemmGrid, 128, 0, stream>>>(Qin, Wqb, bq, Qp, M, D, D, 0.125f);
    gemm_xwt_wmma<true><<<gemmGrid, 128, 0, stream>>>(Kin, Wkb, bk, Kp, M, D, D, 1.0f);
    gemm_xwt_wmma<true><<<gemmGrid, 128, 0, stream>>>(Vin, Wvb, bv, Vp, M, D, D, 1.0f);

    // 3) causal flash attention -> Hcat (bf16)
    flash_attn_wmma<<<dim3(SEQ / 64, N_HEADS, BATCH), 128, 0, stream>>>(Qp, Kp, Vp, Hc);

    // 4) output projection -> f32 d_out
    gemm_xwt_wmma<false><<<gemmGrid, 128, 0, stream>>>(Hc, Wob, bo, d_out, M, D, D, 1.0f);
}
